// patchMatch_74775380623730
// MI455X (gfx1250) — matrix operand order repaired and checked by reading, not compiled, not run on hardware
//
#include <hip/hip_runtime.h>
#include <hip/hip_bf16.h>

typedef __attribute__((ext_vector_type(16))) _Float16 v16h;
typedef __attribute__((ext_vector_type(8)))  _Float16 v8h;
typedef __attribute__((ext_vector_type(8)))  float    v8f;

#define B_  2
#define C_  32
#define H_  160
#define W_  320
#define S_  8
#define F_  3
#define D_  192            // logical disparity band width
#define DP_ 224            // padded band row (stores at d+16, always in [1,223])
#define DOFF_ 16
#define MAX_DISP_F 192.0f
#define TEMP_F 7.0f

#define MTILES_ (W_ / 16)          // 20
#define NTILES_TOTAL_ 182          // sum over m of (min(m,12)+1)
#define LROW_ 40                   // padded LDS row stride in halves (80 B)

// ---------------------------------------------------------------------------
// Phase 0: stratified random init of disparity candidates (B,S,H,W)
// ---------------------------------------------------------------------------
__global__ __launch_bounds__(256) void init_cand_kernel(
    const float* __restrict__ rand_init, float* __restrict__ cand, int total)
{
    int i = blockIdx.x * blockDim.x + threadIdx.x;
    if (i >= total) return;
    int s = (i / (H_ * W_)) % S_;
    const float step = MAX_DISP_F / (float)S_;       // 24.0
    float v = (float)s * step + rand_init[i] * step; // MIN_DISP == 0
    v = fminf(fmaxf(v, 0.0f), MAX_DISP_F - 1.0f);
    cand[i] = v;
}

// ---------------------------------------------------------------------------
// Phase 1: banded correlation volume via WMMA, one 8-wave block per (b,h) row.
//   Stage f16 feature rows in LDS ([w][c] layout, padded row), then each wave
//   processes band tiles with ds_load_b128 fragment reads and branch-free
//   stores (single base + immediate offsets) into the padded corr band.
// ---------------------------------------------------------------------------
__global__ __launch_bounds__(256) void corr_wmma_kernel(
    const float* __restrict__ L, const float* __restrict__ R,
    float* __restrict__ corr)
{
    __shared__ _Float16 shL[W_ * LROW_];   // 25,600 B
    __shared__ _Float16 shR[W_ * LROW_];   // 25,600 B

    const int h = blockIdx.x;
    const int b = blockIdx.y;
    const int tid = threadIdx.x;

    const size_t HW = (size_t)H_ * W_;
    const float* Lb = L + ((size_t)b * C_ * H_ + h) * W_;   // + c*HW + w
    const float* Rb = R + ((size_t)b * C_ * H_ + h) * W_;

    // ---- cooperative f32 -> f16 staging, coalesced along w
    for (int e = tid; e < C_ * W_; e += 256) {
        int c = e / W_;
        int w = e - c * W_;
        shL[w * LROW_ + c] = (_Float16)Lb[(size_t)c * HW + w];
        shR[w * LROW_ + c] = (_Float16)Rb[(size_t)c * HW + w];
    }
    __syncthreads();

    const int wave = tid >> 5;
    const int lane = tid & 31;
    const int hi   = lane >> 4;
    const int lo   = lane & 15;

    float* crow = corr + (((size_t)b * H_ + h) * W_) * DP_;

    for (int t = wave; t < NTILES_TOTAL_; t += 8) {
        // decode tile id -> (m, n): n in [max(0, m-12), m]
        int m = 0, acc = 0;
        while (true) {
            int cnt = (m < 12) ? (m + 1) : 13;
            if (t < acc + cnt) break;
            acc += cnt;
            ++m;
        }
        int n = ((m - 12 > 0) ? (m - 12) : 0) + (t - acc);

        const int w0  = m * 16;
        const int wp0 = n * 16;

        // ---- A fragment (left): lane (hi,lo) = row M=lo,
        //      halves K in [8hi,8hi+8) then [16+8hi,16+8hi+8)
        union { v16h v; v8h hh[2]; } au;
        const _Float16* ap = &shL[(w0 + lo) * LROW_ + 8 * hi];
        au.hh[0] = *(const v8h*)(ap);
        au.hh[1] = *(const v8h*)(ap + 16);

        // ---- B fragment (right): lane (hi,lo) = col N=lo,
        //      halves K in [8hi,8hi+8) then [16+8hi,16+8hi+8)
        union { v16h v; v8h hh[2]; } bu;
        const _Float16* bp = &shR[(wp0 + lo) * LROW_ + 8 * hi];
        bu.hh[0] = *(const v8h*)(bp);
        bu.hh[1] = *(const v8h*)(bp + 16);

        v8f accv = {};
        accv = __builtin_amdgcn_wmma_f32_16x16x32_f16(
            /*neg_a=*/false, au.v, /*neg_b=*/false, bu.v,
            /*c_mod=*/(short)0, accv, /*reuse_a=*/false, /*reuse_b=*/false);

        // ---- branch-free scatter into padded band:
        //      D element (M = r + 8hi, N = lo); d+16 = (w - wp) + 16 in [1,223].
        //      Consecutive r advance the address by exactly (DP_+1) floats
        //      (900 B), so use ONE per-lane base pointer + immediate offsets.
        const int dbase = (w0 + 8 * hi) - (wp0 + lo) + DOFF_;   // >= 1
        float* outp = crow + (size_t)(w0 + 8 * hi) * DP_ + dbase;
#pragma unroll
        for (int r = 0; r < 8; ++r) {
            outp[r * (DP_ + 1)] = accv[r];
        }
    }
}

// ---------------------------------------------------------------------------
// Phase 2: fused propagate (replication-padded 3-tap shift) + evaluate
//   (softmax over the 3 taps using precomputed correlations).
// ---------------------------------------------------------------------------
__global__ __launch_bounds__(256) void eval_kernel(
    const float* __restrict__ cin, const float* __restrict__ corr,
    float* __restrict__ cout, int axis_h)
{
    int i = blockIdx.x * blockDim.x + threadIdx.x;
    const int total = B_ * S_ * H_ * W_;
    if (i >= total) return;

    const int w = i % W_;
    const int h = (i / W_) % H_;
    const int s = (i / (W_ * H_)) % S_;
    const int b = i / (W_ * H_ * S_);

    const float* crow = corr + (((size_t)b * H_ + h) * W_ + w) * DP_;
    const float scale = TEMP_F / (float)C_;

    float cs[F_], wt[F_];
#pragma unroll
    for (int f = 0; f < F_; ++f) {
        int hh = h, ww = w;
        if (axis_h) {
            hh = h + f - 1; hh = hh < 0 ? 0 : (hh > H_ - 1 ? H_ - 1 : hh);
        } else {
            ww = w + f - 1; ww = ww < 0 ? 0 : (ww > W_ - 1 ? W_ - 1 : ww);
        }
        float c = cin[(((size_t)b * S_ + s) * H_ + hh) * W_ + ww];
        // torch-gather semantics: clamp then truncate (c>=0 so trunc==floor)
        float t = fminf(fmaxf((float)w - c, 0.0f), (float)(W_ - 1));
        int idx = (int)t;
        int d = w - idx;
        d = d < 0 ? 0 : (d > D_ - 1 ? D_ - 1 : d);
        wt[f] = crow[d + DOFF_] * scale;
        cs[f] = c;
    }

    float mx = fmaxf(wt[0], fmaxf(wt[1], wt[2]));
    float e0 = __expf(wt[0] - mx);
    float e1 = __expf(wt[1] - mx);
    float e2 = __expf(wt[2] - mx);
    float inv = 1.0f / (e0 + e1 + e2);
    cout[i] = (cs[0] * e0 + cs[1] * e1 + cs[2] * e2) * inv;
}

// ---------------------------------------------------------------------------
extern "C" void kernel_launch(void* const* d_in, const int* in_sizes, int n_in,
                              void* d_out, int out_size, void* d_ws, size_t ws_size,
                              hipStream_t stream)
{
    const float* left      = (const float*)d_in[0];   // (B,C,H,W) f32
    const float* right     = (const float*)d_in[1];   // (B,C,H,W) f32
    const float* rand_init = (const float*)d_in[2];   // (B,S,H,W) f32
    float* out = (float*)d_out;                       // (B,S,H,W) f32

    char* ws = (char*)d_ws;
    float* corr = (float*)ws;                                      // 91.8 MB
    size_t corr_elems = (size_t)B_ * H_ * W_ * DP_;
    float* c0 = (float*)(ws + corr_elems * sizeof(float));         // 3.3 MB
    float* c1 = c0 + (size_t)B_ * S_ * H_ * W_;                    // 3.3 MB

    const int total = B_ * S_ * H_ * W_;
    const int blk = 256;
    const int grd = (total + blk - 1) / blk;

    // Phase 0: candidates
    init_cand_kernel<<<grd, blk, 0, stream>>>(rand_init, c0, total);

    // Phase 1: banded correlation volume (WMMA), computed once, reused 4x
    dim3 cg(H_, B_);
    corr_wmma_kernel<<<cg, 256, 0, stream>>>(left, right, corr);

    // Phase 2: iter 1 horizontal, vertical; iter 2 horizontal, vertical
    eval_kernel<<<grd, blk, 0, stream>>>(c0, corr, c1, /*axis_h=*/0);
    eval_kernel<<<grd, blk, 0, stream>>>(c1, corr, c0, /*axis_h=*/1);
    eval_kernel<<<grd, blk, 0, stream>>>(c0, corr, c1, /*axis_h=*/0);
    eval_kernel<<<grd, blk, 0, stream>>>(c1, corr, out, /*axis_h=*/1);
}
